// NeuronalAttention_51290499449087
// MI455X (gfx1250) — matrix operand order repaired and checked by reading, not compiled, not run on hardware
//
#include <hip/hip_runtime.h>

#define HID 1024
#define NH  16
#define HD  64
#define BB  4
#define SS  1024

typedef __bf16 bf16_t;
typedef bf16_t v16bf __attribute__((ext_vector_type(16)));
typedef float  v8f   __attribute__((ext_vector_type(8)));

union FragU { v16bf bf; unsigned u[8]; };

#if defined(__gfx1250__) && __has_builtin(__builtin_amdgcn_global_load_async_to_lds_b128)
#define USE_ASYNC_LDS 1
#else
#define USE_ASYNC_LDS 0
#endif

#if USE_ASYNC_LDS
// Builtin signature (from hipcc diagnostic): param0 = int4 AS1*, param1 (expected) = int4 AS3*.
typedef int v4i __attribute__((__vector_size__(16)));
typedef __attribute__((address_space(1))) v4i* gbl4_t;
typedef __attribute__((address_space(3))) v4i* lds4_t;
#endif

__device__ __forceinline__ void wait_asynccnt0() {
#if __has_builtin(__builtin_amdgcn_s_wait_asynccnt)
  __builtin_amdgcn_s_wait_asynccnt(0);
#else
  asm volatile("s_wait_asynccnt 0x0" ::: "memory");
#endif
}

__device__ __forceinline__ unsigned short f2bf(float f) {
  union { float f; unsigned u; } v; v.f = f;
  unsigned r = v.u + 0x7FFFu + ((v.u >> 16) & 1u);
  return (unsigned short)(r >> 16);
}

__device__ __forceinline__ float sigmoidf_(float x) {
  return 1.0f / (1.0f + __expf(-x));
}

// ---------------------------------------------------------------------------
// f32 -> bf16 convert
// ---------------------------------------------------------------------------
__global__ __launch_bounds__(256) void cvt_bf16_kernel(const float* __restrict__ s,
                                                       unsigned short* __restrict__ d, int n) {
  int i = blockIdx.x * blockDim.x + threadIdx.x;
  if (i < n) d[i] = f2bf(s[i]);
}

// ---------------------------------------------------------------------------
// Pack gate input: gi[r, 0:1024] = bf16(ctx), gi[r, 1024:2048] = bf16(mem)
// ---------------------------------------------------------------------------
__global__ __launch_bounds__(256) void pack_gate_kernel(const float* __restrict__ ctx,
                                                        const float* __restrict__ mem,
                                                        unsigned short* __restrict__ gi, int n) {
  int i = blockIdx.x * blockDim.x + threadIdx.x;
  if (i >= n) return;
  int r = i >> 10, j = i & 1023;
  gi[(size_t)r * 2048 + j]        = f2bf(ctx[i]);
  gi[(size_t)r * 2048 + 1024 + j] = f2bf(mem[i]);
}

// ---------------------------------------------------------------------------
// Generic WMMA GEMM: C[M,N] = A[M,K](bf16) * W[N,K]^T(bf16) + bias, epilogues:
//   EPI 0: store bf16 C                              (Q, K projections)
//   EPI 1: C *= sigmoid(aux); store bf16 TRANSPOSED  (V projection -> Vt[b,h,d,s])
//   EPI 2: store bf16 aux * sigmoid(C)               (dendritic gate)
//   EPI 3: store f32 C                               (output projection)
// Block tile 64x128, 8 waves of 32x32, K-chunk 32.
// LDS rows padded to 40 shorts (80B): 16B-aligned row starts for b128 staging,
// 20-dword gather stride -> conflict-free across 16-lane fragment groups.
// ---------------------------------------------------------------------------
template <int EPI>
__global__ __launch_bounds__(256)
void wmma_gemm_kernel(const unsigned short* __restrict__ A,
                      const unsigned short* __restrict__ W,
                      const float* __restrict__ bias,
                      const float* __restrict__ aux,
                      void* __restrict__ outp,
                      int M, int N, int K) {
  __shared__ unsigned short lA[64][40];
  __shared__ unsigned short lB[128][40];
  const int t    = threadIdx.x;
  const int lane = t & 31;
  const int wave = t >> 5;
  const int half = lane >> 4;
  const int l16  = lane & 15;
  const int wm   = wave >> 2;   // 0..1
  const int wn   = wave & 3;    // 0..3
  const int blockM = blockIdx.y * 64;
  const int blockN = blockIdx.x * 128;

  const int rowA = t >> 2, cbA = (t & 3) * 8;   // 64 rows x 4 x 8 shorts
  const int rowB = t >> 1, cbB = (t & 1) * 16;  // 128 rows x 2 x 16 shorts
  const unsigned short* gA = A + (size_t)(blockM + rowA) * K + cbA;
  const unsigned short* gB = W + (size_t)(blockN + rowB) * K + cbB;

  v8f c[2][2];
#pragma unroll
  for (int i = 0; i < 2; i++)
#pragma unroll
    for (int j = 0; j < 2; j++)
#pragma unroll
      for (int e = 0; e < 8; e++) c[i][j][e] = 0.0f;

  for (int kt = 0; kt < K; kt += 32) {
#if USE_ASYNC_LDS
    // Async DMA global -> LDS (ASYNCcnt), one/two b128 per thread.
    __builtin_amdgcn_global_load_async_to_lds_b128(
        (gbl4_t)(gA + kt), (lds4_t)&lA[rowA][cbA], 0, 0);
    __builtin_amdgcn_global_load_async_to_lds_b128(
        (gbl4_t)(gB + kt), (lds4_t)&lB[rowB][cbB], 0, 0);
    __builtin_amdgcn_global_load_async_to_lds_b128(
        (gbl4_t)(gB + kt + 8), (lds4_t)&lB[rowB][cbB + 8], 0, 0);
    wait_asynccnt0();
#else
    *(uint4*)&lA[rowA][cbA]     = *(const uint4*)(gA + kt);
    *(uint4*)&lB[rowB][cbB]     = *(const uint4*)(gB + kt);
    *(uint4*)&lB[rowB][cbB + 8] = *(const uint4*)(gB + kt + 8);
#endif
    // Prefetch next K tile while this chunk computes.
    if (kt + 32 < K) {
      __builtin_prefetch(gA + kt + 32, 0, 1);
      __builtin_prefetch(gB + kt + 32, 0, 1);
    }
    __syncthreads();

    FragU a[2], b[2];
#pragma unroll
    for (int i = 0; i < 2; i++) {
      int mo = wm * 32 + i * 16 + l16;
#pragma unroll
      for (int v = 0; v < 8; v++) {
        int k0 = ((v >> 2) << 4) + ((v & 3) << 1) + (half << 3);  // A layout
        a[i].u[v] = *(const unsigned*)&lA[mo][k0];
      }
    }
#pragma unroll
    for (int j = 0; j < 2; j++) {
      int no = wn * 32 + j * 16 + l16;
#pragma unroll
      for (int v = 0; v < 8; v++) {
        int k0 = ((v >> 2) << 4) + ((v & 3) << 1) + (half << 3);  // B layout
        b[j].u[v] = *(const unsigned*)&lB[no][k0];
      }
    }
#pragma unroll
    for (int i = 0; i < 2; i++)
#pragma unroll
      for (int j = 0; j < 2; j++)
        c[i][j] = __builtin_amdgcn_wmma_f32_16x16x32_bf16(
            false, a[i].bf, false, b[j].bf, (short)0, c[i][j], false, false);
    __syncthreads();
  }

#pragma unroll
  for (int i = 0; i < 2; i++) {
#pragma unroll
    for (int j = 0; j < 2; j++) {
      int gn = blockN + wn * 32 + j * 16 + l16;
      float bval = bias ? bias[gn] : 0.0f;
#pragma unroll
      for (int v = 0; v < 8; v++) {
        int gm = blockM + wm * 32 + i * 16 + v + half * 8;
        float acc = c[i][j][v] + bval;
        if (EPI == 0) {
          ((unsigned short*)outp)[(size_t)gm * N + gn] = f2bf(acc);
        } else if (EPI == 1) {
          float g   = sigmoidf_(aux[(size_t)gm * N + gn]);
          int bidx  = gm >> 10, s = gm & 1023;
          int h     = gn >> 6,  d = gn & 63;
          ((unsigned short*)outp)[(((size_t)(bidx * NH + h)) * HD + d) * SS + s] = f2bf(acc * g);
        } else if (EPI == 2) {
          float g = sigmoidf_(acc);
          ((unsigned short*)outp)[(size_t)gm * N + gn] =
              f2bf(aux[(size_t)gm * N + gn] * g);
        } else {
          ((float*)outp)[(size_t)gm * N + gn] = acc;
        }
      }
    }
  }
}

// ---------------------------------------------------------------------------
// Fused neuronal flash attention.
// Grid: BB*NH*8 blocks x 256 threads; each wave owns a 16-row q tile.
// Per 32-wide kv chunk: 4 WMMA (QK), 4 WMMA (spike), decay/spike epilogue,
// online softmax w/ shfl_xor row reductions, 4 WMMA (P*V) from pre-transposed
// gated Vt[b,h,d,s].
// ---------------------------------------------------------------------------
__global__ __launch_bounds__(256)
void attention_kernel(const unsigned short* __restrict__ Q,
                      const unsigned short* __restrict__ Km,
                      const unsigned short* __restrict__ Sp,
                      const unsigned short* __restrict__ Vt,
                      const float* __restrict__ tau,
                      float* __restrict__ ctx) {
  __shared__ unsigned short lP[8][16][34];
  const int lane = threadIdx.x & 31;
  const int wave = threadIdx.x >> 5;
  const int half = lane >> 4;
  const int l16  = lane & 15;
  int blk  = blockIdx.x;
  int qgrp = blk & 7;
  int r    = blk >> 3;
  int h    = r & (NH - 1);
  int b    = r >> 4;
  int qbase = (qgrp * 8 + wave) * 16;
  float inv_tau = 1.0f / tau[h];
  const float scale = 0.125f;  // 1/sqrt(64)

  FragU qa[2], sa[2];
  {
    size_t rowbase = ((size_t)(b * SS) + qbase + l16) * HID + h * HD;
#pragma unroll
    for (int kc = 0; kc < 2; kc++)
#pragma unroll
      for (int v = 0; v < 8; v++) {
        int k0 = ((v >> 2) << 4) + ((v & 3) << 1) + (half << 3);
        qa[kc].u[v] = *(const unsigned*)(Q + rowbase + kc * 32 + k0);
        sa[kc].u[v] = *(const unsigned*)(Sp + rowbase + kc * 32 + k0);
      }
  }

  float mrun[8], lrun[8];
  v8f acc[4];
#pragma unroll
  for (int v = 0; v < 8; v++) { mrun[v] = -1e30f; lrun[v] = 0.0f; }
#pragma unroll
  for (int cdx = 0; cdx < 4; cdx++)
#pragma unroll
    for (int v = 0; v < 8; v++) acc[cdx][v] = 0.0f;

  for (int kv = 0; kv < SS; kv += 32) {
    v8f sc[2], sp[2];
#pragma unroll
    for (int tt = 0; tt < 2; tt++)
#pragma unroll
      for (int v = 0; v < 8; v++) { sc[tt][v] = 0.0f; sp[tt][v] = 0.0f; }

#pragma unroll
    for (int tt = 0; tt < 2; tt++) {
      size_t colbase = ((size_t)(b * SS) + kv + tt * 16 + l16) * HID + h * HD;
#pragma unroll
      for (int kc = 0; kc < 2; kc++) {
        FragU kb, sb;
#pragma unroll
        for (int v = 0; v < 8; v++) {
          int k0 = ((v >> 2) << 4) + ((v & 3) << 1) + (half << 3);
          kb.u[v] = *(const unsigned*)(Km + colbase + kc * 32 + k0);
          sb.u[v] = *(const unsigned*)(Sp + colbase + kc * 32 + k0);
        }
        sc[tt] = __builtin_amdgcn_wmma_f32_16x16x32_bf16(
            false, qa[kc].bf, false, kb.bf, (short)0, sc[tt], false, false);
        sp[tt] = __builtin_amdgcn_wmma_f32_16x16x32_bf16(
            false, sa[kc].bf, false, sb.bf, (short)0, sp[tt], false, false);
      }
    }

    float p[2][8];
#pragma unroll
    for (int tt = 0; tt < 2; tt++)
#pragma unroll
      for (int v = 0; v < 8; v++) {
        int tq = qbase + v + half * 8;
        int tk = kv + tt * 16 + l16;
        float mask = __expf(-fabsf((float)(tq - tk)) * inv_tau);
        p[tt][v] = sc[tt][v] * scale * mask * (1.0f + sp[tt][v] * scale);
      }

#pragma unroll
    for (int v = 0; v < 8; v++) {
      float cm = fmaxf(p[0][v], p[1][v]);
#pragma unroll
      for (int m = 1; m < 16; m <<= 1) cm = fmaxf(cm, __shfl_xor(cm, m, 32));
      float nm  = fmaxf(mrun[v], cm);
      float fct = __expf(mrun[v] - nm);
      float e0  = __expf(p[0][v] - nm);
      float e1  = __expf(p[1][v] - nm);
      float rs  = e0 + e1;
#pragma unroll
      for (int m = 1; m < 16; m <<= 1) rs += __shfl_xor(rs, m, 32);
      lrun[v] = lrun[v] * fct + rs;
      mrun[v] = nm;
      int row = v + half * 8;
      lP[wave][row][l16]      = f2bf(e0);
      lP[wave][row][16 + l16] = f2bf(e1);
#pragma unroll
      for (int cdx = 0; cdx < 4; cdx++) acc[cdx][v] *= fct;
    }

    asm volatile("s_wait_dscnt 0" ::: "memory");

    FragU pa;
#pragma unroll
    for (int v = 0; v < 8; v++) {
      int k0 = ((v >> 2) << 4) + ((v & 3) << 1) + (half << 3);
      pa.u[v] = *(const unsigned*)&lP[wave][l16][k0];
    }

#pragma unroll
    for (int cdx = 0; cdx < 4; cdx++) {
      FragU vb;
      size_t dbase = (((size_t)(b * NH + h)) * HD + cdx * 16 + l16) * SS + kv;
#pragma unroll
      for (int v = 0; v < 8; v++) {
        int k0 = ((v >> 2) << 4) + ((v & 3) << 1) + (half << 3);
        vb.u[v] = *(const unsigned*)(Vt + dbase + k0);
      }
      acc[cdx] = __builtin_amdgcn_wmma_f32_16x16x32_bf16(
          false, pa.bf, false, vb.bf, (short)0, acc[cdx], false, false);
    }
  }

#pragma unroll
  for (int cdx = 0; cdx < 4; cdx++) {
    int d = cdx * 16 + l16;
#pragma unroll
    for (int v = 0; v < 8; v++) {
      int tq = qbase + v + half * 8;
      ctx[((size_t)(b * SS) + tq) * HID + h * HD + d] = acc[cdx][v] / lrun[v];
    }
  }
}

// ---------------------------------------------------------------------------
extern "C" void kernel_launch(void* const* d_in, const int* in_sizes, int n_in,
                              void* d_out, int out_size, void* d_ws, size_t ws_size,
                              hipStream_t stream) {
  const float* x   = (const float*)d_in[0];
  const float* spk = (const float*)d_in[1];
  const float* mem = (const float*)d_in[2];
  const float* Wq  = (const float*)d_in[3];
  const float* bq  = (const float*)d_in[4];
  const float* Wk  = (const float*)d_in[5];
  const float* bk  = (const float*)d_in[6];
  const float* Wv  = (const float*)d_in[7];
  const float* bv  = (const float*)d_in[8];
  const float* Wo  = (const float*)d_in[9];
  const float* bo  = (const float*)d_in[10];
  const float* tau = (const float*)d_in[11];
  const float* Wg  = (const float*)d_in[12];
  const float* bg  = (const float*)d_in[13];
  float* out = (float*)d_out;

  char* ws = (char*)d_ws;
  size_t off = 0;
  auto alloc = [&](size_t bytes) -> char* {
    char* p = ws + off;
    off += (bytes + 255) & ~(size_t)255;
    return p;
  };
  const size_t NTOK = (size_t)BB * SS;  // 4096 rows
  unsigned short* xb   = (unsigned short*)alloc(NTOK * HID * 2);
  unsigned short* sb   = (unsigned short*)alloc(NTOK * HID * 2);
  unsigned short* wqb  = (unsigned short*)alloc((size_t)HID * HID * 2);
  unsigned short* wkb  = (unsigned short*)alloc((size_t)HID * HID * 2);
  unsigned short* wvb  = (unsigned short*)alloc((size_t)HID * HID * 2);
  unsigned short* wob  = (unsigned short*)alloc((size_t)HID * HID * 2);
  unsigned short* wgb  = (unsigned short*)alloc((size_t)HID * 2 * HID * 2);
  unsigned short* Qb   = (unsigned short*)alloc(NTOK * HID * 2);
  unsigned short* Kb   = (unsigned short*)alloc(NTOK * HID * 2);
  unsigned short* Vt   = (unsigned short*)alloc(NTOK * HID * 2);
  float*          ctx  = (float*)alloc(NTOK * HID * 4);
  unsigned short* gib  = (unsigned short*)alloc(NTOK * 2 * HID * 2);
  unsigned short* cgb  = (unsigned short*)alloc(NTOK * HID * 2);

  const int nAct = (int)(NTOK * HID);        // 4,194,304
  const int nW   = HID * HID;                // 1,048,576
  const int nWg  = HID * 2 * HID;            // 2,097,152

  cvt_bf16_kernel<<<(nAct + 255) / 256, 256, 0, stream>>>(x,   xb,  nAct);
  cvt_bf16_kernel<<<(nAct + 255) / 256, 256, 0, stream>>>(spk, sb,  nAct);
  cvt_bf16_kernel<<<(nW  + 255) / 256, 256, 0, stream>>>(Wq,  wqb, nW);
  cvt_bf16_kernel<<<(nW  + 255) / 256, 256, 0, stream>>>(Wk,  wkb, nW);
  cvt_bf16_kernel<<<(nW  + 255) / 256, 256, 0, stream>>>(Wv,  wvb, nW);
  cvt_bf16_kernel<<<(nW  + 255) / 256, 256, 0, stream>>>(Wo,  wob, nW);
  cvt_bf16_kernel<<<(nWg + 255) / 256, 256, 0, stream>>>(Wg,  wgb, nWg);

  dim3 gGemm(HID / 128, (int)(NTOK / 64));   // (8, 64)

  // Q = x Wq^T + bq ; K = x Wk^T + bk (bf16)
  wmma_gemm_kernel<0><<<gGemm, 256, 0, stream>>>(xb, wqb, bq, nullptr, Qb,
                                                 (int)NTOK, HID, HID);
  wmma_gemm_kernel<0><<<gGemm, 256, 0, stream>>>(xb, wkb, bk, nullptr, Kb,
                                                 (int)NTOK, HID, HID);
  // Vt[b,h,d,s] = (x Wv^T + bv) * sigmoid(membrane), transposed store
  wmma_gemm_kernel<1><<<gGemm, 256, 0, stream>>>(xb, wvb, bv, mem, Vt,
                                                 (int)NTOK, HID, HID);

  // fused attention -> ctx (f32)
  attention_kernel<<<BB * NH * 8, 256, 0, stream>>>(Qb, Kb, sb, Vt, tau, ctx);

  // gate input concat
  pack_gate_kernel<<<(nAct + 255) / 256, 256, 0, stream>>>(ctx, mem, gib, nAct);

  // gated context = ctx * sigmoid(gi Wg^T + bg) (bf16)
  wmma_gemm_kernel<2><<<gGemm, 256, 0, stream>>>(gib, wgb, bg, ctx, cgb,
                                                 (int)NTOK, HID, 2 * HID);

  // out = gated_ctx Wo^T + bo (f32)
  wmma_gemm_kernel<3><<<gGemm, 256, 0, stream>>>(cgb, wob, bo, nullptr, out,
                                                 (int)NTOK, HID, HID);
}
